// _NonLocalBlockND_7516192768414
// MI455X (gfx1250) — compile-verified, matrix operand order checked by reading, not run on hardware
//
#include <hip/hip_runtime.h>

typedef unsigned short u16;
typedef __attribute__((ext_vector_type(16))) __bf16 v16bf;
typedef __attribute__((ext_vector_type(8)))  float  v8f;
typedef __attribute__((ext_vector_type(4)))  int    v4i;

#define LDT 40   // padded LDS row stride (elements): keeps b128 stores and dword pair-gathers aligned

// ---------- optional CDNA5 async global->LDS path (guarded; falls back to ld+st) ----------
#if defined(__has_builtin)
#if __has_builtin(__builtin_amdgcn_global_load_async_to_lds_b128)
#define HAS_ASYNC_LDS 1
#endif
#if __has_builtin(__builtin_amdgcn_s_wait_asynccnt)
#define HAS_WAIT_ASYNC 1
#endif
#endif

__device__ __forceinline__ void cp16(u16* l, const u16* g) {
#if defined(HAS_ASYNC_LDS)
  __builtin_amdgcn_global_load_async_to_lds_b128(
      (__attribute__((address_space(1))) v4i*)(u16*)g,
      (__attribute__((address_space(3))) v4i*)l, 0, 0);
#else
  *(uint4*)l = *(const uint4*)g;
#endif
}

__device__ __forceinline__ void async_wait() {
#if defined(HAS_ASYNC_LDS)
#if defined(HAS_WAIT_ASYNC)
  __builtin_amdgcn_s_wait_asynccnt(0);
#else
  asm volatile("s_wait_asynccnt 0x0" ::: "memory");
#endif
#endif
}

// ---------- bf16 helpers ----------
__device__ __forceinline__ u16 f2bf(float f) {
  unsigned u = __float_as_uint(f);
  return (u16)((u + 0x7FFFu + ((u >> 16) & 1u)) >> 16);  // RNE
}
__device__ __forceinline__ float bf2f(u16 h) {
  return __uint_as_float(((unsigned)h) << 16);
}

// ---------- WMMA fragment gather ----------
// 16-bit A (16x32) layout: lanes 0-15 = rows M=0..15 (K 0..7 in VGPR0-3, 16..23 in VGPR4-7),
// lanes 16-31 same rows, K+8 / K+24.  B (32x16) mirrors with lane = column N.
__device__ __forceinline__ v16bf ldfrag(const u16* p, int hi) {
  union { v16bf v; unsigned u[8]; } f;
#pragma unroll
  for (int g = 0; g < 8; ++g) {
    int kb = ((g < 4) ? 2 * g : 16 + 2 * (g - 4)) + hi * 8;  // even k base
    f.u[g] = *(const unsigned*)(p + kb);                      // 2 bf16 per dword
  }
  return f.v;
}

__device__ __forceinline__ v8f wmma_bf16(v16bf a, v16bf b, v8f c) {
  return __builtin_amdgcn_wmma_f32_16x16x32_bf16(false, a, false, b, (short)0, c, false, false);
}

// ---------- tile staging (one K-step) ----------
// A tile 256x32 via 16B copies (async-to-LDS when available); B tile 32x16.
__device__ __forceinline__ void stage_tiles(
    const u16* __restrict__ A, int ldA,
    const u16* __restrict__ B, int ldB,
    int mBase, int nBase, int kb, bool transB,
    u16* As, u16* Bs, int t)
{
  const int r = t >> 2, c = (t & 3) * 8;
#pragma unroll
  for (int j = 0; j < 4; ++j) {
    const int row = r + 64 * j;
    cp16(As + row * LDT + c, A + (size_t)(mBase + row) * ldA + kb + c);
  }
  if (transB) {  // B[k][n] -> LDS [n][k]; 2 cols per thread, branch-free
    const int k = t >> 3, n2 = (t & 7) * 2;
    const u16* src = B + (size_t)(kb + k) * ldB + nBase + n2;
    const u16 b0 = src[0], b1 = src[1];
    Bs[n2 * LDT + k]       = b0;
    Bs[(n2 + 1) * LDT + k] = b1;
  } else {       // B[n][k] -> LDS [n][k]; dword copy
    const int n = t >> 4, c2 = (t & 15) * 2;
    *(unsigned*)(Bs + n * LDT + c2) =
        *(const unsigned*)(B + (size_t)(nBase + n) * ldB + kb + c2);
  }
}

// ---------- shared GEMM core ----------
// Block: 256 threads = 8 waves stacked in M. Block tile 256(M) x 16(N), K-step 32.
// Wave w computes rows [w*32, w*32+32): two accumulators sharing one B fragment.
// Double-buffered LDS: one barrier per K-step; stage k+1 overlaps compute of k.
// All dims exact multiples (M%256==0, N%16==0, K%32==0) -> zero guards.
__device__ __forceinline__ void gemm_core(
    const u16* __restrict__ A, int ldA,
    const u16* __restrict__ B, int ldB,
    int mBase, int nBase, int K, bool transB,
    u16* As, u16* Bs, v8f& acc0, v8f& acc1)   // As: 2*256*LDT, Bs: 2*16*LDT
{
  const int t    = threadIdx.x;
  const int lane = t & 31;
  const int wave = t >> 5;
  const int mo   = lane & 15, hi = lane >> 4;
  const int steps = K >> 5;

  stage_tiles(A, ldA, B, ldB, mBase, nBase, 0, transB, As, Bs, t);
  for (int i = 0; i < steps; ++i) {
    u16* Ac = As + (i & 1) * (256 * LDT);
    u16* Bc = Bs + (i & 1) * (16 * LDT);
    async_wait();       // stage i resident in LDS (wave's own async ops)
    __syncthreads();    // all waves staged i; all waves done reading buffer (i+1)&1
    if (i + 1 < steps)
      stage_tiles(A, ldA, B, ldB, mBase, nBase, (i + 1) * 32, transB,
                  As + ((i + 1) & 1) * (256 * LDT), Bs + ((i + 1) & 1) * (16 * LDT), t);
    if (i + 2 < steps)  // prefetch K-tile i+2 (global_prefetch_b8)
      __builtin_prefetch(A + (size_t)(mBase + (t >> 2)) * ldA + (i + 2) * 32 + (t & 3) * 8, 0, 3);
    const v16bf bf = ldfrag(Bc + mo * LDT, hi);
    const v16bf a0 = ldfrag(Ac + (wave * 32 + mo) * LDT, hi);
    const v16bf a1 = ldfrag(Ac + (wave * 32 + 16 + mo) * LDT, hi);
    acc0 = wmma_bf16(a0, bf, acc0);
    acc1 = wmma_bf16(a1, bf, acc1);
  }
}

// ---------- elementwise kernels ----------
__global__ void k_cvt_x(const float* __restrict__ x, u16* __restrict__ xb) {
  size_t i = ((size_t)blockIdx.x * 256 + threadIdx.x) * 4;
#pragma unroll
  for (int j = 0; j < 4; ++j) xb[i + j] = f2bf(x[i + j]);
}

__global__ void k_cvt_w(const float* __restrict__ wth, const float* __restrict__ wg,
                        const float* __restrict__ wph, const float* __restrict__ bth,
                        const float* __restrict__ bg,  const float* __restrict__ bph,
                        const float* __restrict__ wW,
                        u16* __restrict__ Wpack, u16* __restrict__ wWb,
                        float* __restrict__ biasPack) {
  int i = blockIdx.x * 256 + threadIdx.x;
  if (i < 768 * 512) {  // rows 0-255 theta, 256-511 g, 512-767 phi
    int r = i >> 9, c = i & 511;
    const float* s = (r < 256) ? wth : (r < 512) ? wg : wph;
    Wpack[i] = f2bf(s[(r & 255) * 512 + c]);
  }
  if (i < 512 * 256) wWb[i] = f2bf(wW[i]);
  if (i < 768) {
    const float* s = (i < 256) ? bth : (i < 512) ? bg : bph;
    biasPack[i] = s[i & 255];
  }
}

__global__ void k_init(float* a, float* b) {
  int i = blockIdx.x * 256 + threadIdx.x;
  if (i < 512) { a[i] = 0.f; b[i] = 0.f; }
}

__global__ void k_pool(const u16* __restrict__ gfull, const u16* __restrict__ phifull,
                       u16* __restrict__ gpool, u16* __restrict__ phipool) {
  int i = blockIdx.x * 256 + threadIdx.x;
  const int half = 64 * 256 * 196;
  const u16* src; u16* dst; int j = i;
  if (i < half) { src = gfull;   dst = gpool; }
  else          { src = phifull; dst = phipool; j -= half; }
  int p = j % 196, nc = j / 196;              // nc = n*256 + c
  const u16* r = src + (size_t)nc * 784;
  int ph = p / 14, pw = p % 14;
  int q = (2 * ph) * 28 + 2 * pw;
  float a = bf2f(r[q]),      b = bf2f(r[q + 1]);
  float c = bf2f(r[q + 28]), d = bf2f(r[q + 29]);
  dst[j] = f2bf(fmaxf(fmaxf(a, b), fmaxf(c, d)));
}

// ---------- GEMM kernels ----------
// proj: per image, P[768,784] = Wpack[768,512] @ xb_n[512,784] + biasPack; route rows to theta/g/phi
__global__ void __launch_bounds__(256)
k_gemm_proj(const u16* __restrict__ Wpack, const u16* __restrict__ xb,
            const float* __restrict__ biasPack,
            u16* __restrict__ th, u16* __restrict__ gf, u16* __restrict__ pf) {
  __shared__ u16 As[2 * 256 * LDT];
  __shared__ u16 Bs[2 * 16 * LDT];
  const int img = blockIdx.z;
  const int mBase = blockIdx.x * 256, nBase = blockIdx.y * 16;
  v8f acc0 = {0.f,0.f,0.f,0.f,0.f,0.f,0.f,0.f};
  v8f acc1 = {0.f,0.f,0.f,0.f,0.f,0.f,0.f,0.f};
  gemm_core(Wpack, 512, xb + (size_t)img * 512 * 784, 784,
            mBase, nBase, 512, true, As, Bs, acc0, acc1);
  const int lane = threadIdx.x & 31, wave = threadIdx.x >> 5;
  const int mo = lane & 15, hi = lane >> 4;
  const int n = nBase + mo;
#pragma unroll
  for (int half = 0; half < 2; ++half) {
    const v8f& a = half ? acc1 : acc0;
#pragma unroll
    for (int v = 0; v < 8; ++v) {
      int m = mBase + wave * 32 + half * 16 + v + 8 * hi;
      u16 o = f2bf(a[v] + biasPack[m]);
      size_t off = (size_t)img * 256 * 784 + (size_t)(m & 255) * 784 + n;
      if (m < 256) th[off] = o; else if (m < 512) gf[off] = o; else pf[off] = o;
    }
  }
}

// S: per group b, S[256,256] = Phi_b[256,1568] @ G_b[256,1568]^T; store S^T * (1/1568) in bf16
__global__ void __launch_bounds__(256)
k_gemm_S(const u16* __restrict__ phip, const u16* __restrict__ gp, u16* __restrict__ St) {
  __shared__ u16 As[2 * 256 * LDT];
  __shared__ u16 Bs[2 * 16 * LDT];
  const int b = blockIdx.z;
  const int mBase = blockIdx.x * 256, nBase = blockIdx.y * 16;
  v8f acc0 = {0.f,0.f,0.f,0.f,0.f,0.f,0.f,0.f};
  v8f acc1 = {0.f,0.f,0.f,0.f,0.f,0.f,0.f,0.f};
  gemm_core(phip + (size_t)b * 401408, 1568, gp + (size_t)b * 401408, 1568,
            mBase, nBase, 1568, false, As, Bs, acc0, acc1);
  const int lane = threadIdx.x & 31, wave = threadIdx.x >> 5;
  const int mo = lane & 15, hi = lane >> 4;
  const int n = nBase + mo;
#pragma unroll
  for (int half = 0; half < 2; ++half) {
    const v8f& a = half ? acc1 : acc0;
#pragma unroll
    for (int v = 0; v < 8; ++v) {
      int m = mBase + wave * 32 + half * 16 + v + 8 * hi;
      St[(size_t)b * 65536 + (size_t)n * 256 + m] = f2bf(a[v] * (1.0f / 1568.0f));
    }
  }
}

// Y: per group b, Y[256,6272] = St_b[256,256] @ Theta_b[256,6272]  (raw layout == [64,256,784])
__global__ void __launch_bounds__(256)
k_gemm_Y(const u16* __restrict__ St, const u16* __restrict__ th, u16* __restrict__ ybuf) {
  __shared__ u16 As[2 * 256 * LDT];
  __shared__ u16 Bs[2 * 16 * LDT];
  const int b = blockIdx.z;
  const int mBase = blockIdx.x * 256, nBase = blockIdx.y * 16;
  v8f acc0 = {0.f,0.f,0.f,0.f,0.f,0.f,0.f,0.f};
  v8f acc1 = {0.f,0.f,0.f,0.f,0.f,0.f,0.f,0.f};
  gemm_core(St + (size_t)b * 65536, 256, th + (size_t)b * 1605632, 6272,
            mBase, nBase, 256, true, As, Bs, acc0, acc1);
  const int lane = threadIdx.x & 31, wave = threadIdx.x >> 5;
  const int mo = lane & 15, hi = lane >> 4;
  const int n = nBase + mo;
#pragma unroll
  for (int half = 0; half < 2; ++half) {
    const v8f& a = half ? acc1 : acc0;
#pragma unroll
    for (int v = 0; v < 8; ++v) {
      int m = mBase + wave * 32 + half * 16 + v + 8 * hi;
      ybuf[(size_t)b * 1605632 + (size_t)m * 6272 + n] = f2bf(a[v]);
    }
  }
}

// Z: per image, Z[512,784] = wWb[512,256] @ Y_n[256,784] + bW; fp32 to d_out scratch +
// per-channel sum / sumsq via half-wave shuffle reduction + atomics.
__global__ void __launch_bounds__(256)
k_gemm_Z(const u16* __restrict__ wWb, const u16* __restrict__ ybuf,
         const float* __restrict__ bW, float* __restrict__ z,
         float* __restrict__ stat_s, float* __restrict__ stat_s2) {
  __shared__ u16 As[2 * 256 * LDT];
  __shared__ u16 Bs[2 * 16 * LDT];
  const int img = blockIdx.z;
  const int mBase = blockIdx.x * 256, nBase = blockIdx.y * 16;
  v8f acc0 = {0.f,0.f,0.f,0.f,0.f,0.f,0.f,0.f};
  v8f acc1 = {0.f,0.f,0.f,0.f,0.f,0.f,0.f,0.f};
  gemm_core(wWb, 256, ybuf + (size_t)img * 200704, 784,
            mBase, nBase, 256, true, As, Bs, acc0, acc1);
  const int lane = threadIdx.x & 31, wave = threadIdx.x >> 5;
  const int mo = lane & 15, hi = lane >> 4;
  const int n = nBase + mo;
#pragma unroll
  for (int half = 0; half < 2; ++half) {
    const v8f& a = half ? acc1 : acc0;
#pragma unroll
    for (int v = 0; v < 8; ++v) {
      int m = mBase + wave * 32 + half * 16 + v + 8 * hi;
      float val = a[v] + bW[m];
      z[(size_t)img * 401408 + (size_t)m * 784 + n] = val;
      float s1 = val, s2 = val * val;
#pragma unroll
      for (int d = 1; d < 16; d <<= 1) {  // reduce within each 16-lane half (same m)
        s1 += __shfl_xor(s1, d, 32);
        s2 += __shfl_xor(s2, d, 32);
      }
      if (mo == 0) {
        atomicAdd(&stat_s[m], s1);
        atomicAdd(&stat_s2[m], s2);
      }
    }
  }
}

__global__ void k_stats(const float* __restrict__ s, const float* __restrict__ s2,
                        const float* __restrict__ gamma, const float* __restrict__ beta,
                        float* __restrict__ scale, float* __restrict__ shift) {
  int c = blockIdx.x * 256 + threadIdx.x;
  if (c < 512) {
    const float invN = 1.0f / 50176.0f;  // 64*784
    float m = s[c] * invN;
    float v = s2[c] * invN - m * m;      // biased variance
    float sc = gamma[c] * rsqrtf(v + 1e-5f);
    scale[c] = sc;
    shift[c] = beta[c] - m * sc;
  }
}

__global__ void k_bnres(float* __restrict__ out, const float* __restrict__ x,
                        const float* __restrict__ scale, const float* __restrict__ shift) {
  size_t i = ((size_t)blockIdx.x * 256 + threadIdx.x) * 4;
  int c = (int)((i / 784) % 512);  // 784 % 4 == 0 -> quad shares channel
  float sc = scale[c], sh = shift[c];
#pragma unroll
  for (int j = 0; j < 4; ++j) {
    float zv = out[i + j];
    out[i + j] = zv * sc + sh + x[i + j];
  }
}

// ---------- host launch ----------
extern "C" void kernel_launch(void* const* d_in, const int* in_sizes, int n_in,
                              void* d_out, int out_size, void* d_ws, size_t ws_size,
                              hipStream_t stream) {
  const float* x   = (const float*)d_in[0];
  const float* wg  = (const float*)d_in[1];
  const float* bg  = (const float*)d_in[2];
  const float* wth = (const float*)d_in[3];
  const float* bth = (const float*)d_in[4];
  const float* wph = (const float*)d_in[5];
  const float* bph = (const float*)d_in[6];
  const float* wW  = (const float*)d_in[7];
  const float* bW  = (const float*)d_in[8];
  const float* gamma = (const float*)d_in[9];
  const float* beta  = (const float*)d_in[10];
  float* out = (float*)d_out;

  // workspace carve-up (all sizes 256B-aligned)
  size_t off = 0;
  auto alloc = [&](size_t bytes) -> void* {
    void* p = (char*)d_ws + off;
    off += (bytes + 255) & ~(size_t)255;
    return p;
  };
  u16*   xb       = (u16*)  alloc((size_t)25690112 * 2);  // x bf16 [64,512,784]
  u16*   Wpack    = (u16*)  alloc((size_t)768 * 512 * 2);
  u16*   wWb      = (u16*)  alloc((size_t)512 * 256 * 2);
  float* biasPack = (float*)alloc(768 * 4);
  u16*   thbuf    = (u16*)  alloc((size_t)12845056 * 2);  // theta  [64,256,784]
  u16*   gfull    = (u16*)  alloc((size_t)12845056 * 2);
  u16*   phifull  = (u16*)  alloc((size_t)12845056 * 2);
  u16*   gpool    = (u16*)  alloc((size_t)3211264 * 2);   // [64,256,196]
  u16*   phipool  = (u16*)  alloc((size_t)3211264 * 2);
  u16*   St       = (u16*)  alloc((size_t)524288 * 2);    // S^T bf16 [8,256,256]
  u16*   ybuf     = (u16*)  alloc((size_t)12845056 * 2);  // y bf16 [64,256,784]
  float* stat_s   = (float*)alloc(512 * 4);
  float* stat_s2  = (float*)alloc(512 * 4);
  float* scaleA   = (float*)alloc(512 * 4);
  float* shiftA   = (float*)alloc(512 * 4);
  if (off > ws_size) return;  // workspace too small; bail deterministically

  (void)in_sizes; (void)n_in; (void)out_size;

  k_cvt_x<<<25088, 256, 0, stream>>>(x, xb);
  k_cvt_w<<<1536, 256, 0, stream>>>(wth, wg, wph, bth, bg, bph, wW, Wpack, wWb, biasPack);
  k_init<<<2, 256, 0, stream>>>(stat_s, stat_s2);

  k_gemm_proj<<<dim3(3, 49, 64), 256, 0, stream>>>(Wpack, xb, biasPack, thbuf, gfull, phifull);
  k_pool<<<25088, 256, 0, stream>>>(gfull, phifull, gpool, phipool);
  k_gemm_S<<<dim3(1, 16, 8), 256, 0, stream>>>(phipool, gpool, St);
  k_gemm_Y<<<dim3(1, 392, 8), 256, 0, stream>>>(St, thbuf, ybuf);
  k_gemm_Z<<<dim3(2, 49, 64), 256, 0, stream>>>(wWb, ybuf, bW, out, stat_s, stat_s2);
  k_stats<<<2, 256, 0, stream>>>(stat_s, stat_s2, gamma, beta, scaleA, shiftA);
  k_bnres<<<25088, 256, 0, stream>>>(out, x, scaleA, shiftA);
}
